// TinyCausalSelfAttention_41901700940048
// MI455X (gfx1250) — compile-verified, matrix operand order checked by reading, not run on hardware
//
#include <hip/hip_runtime.h>
#include <hip/hip_bf16.h>
#include <stdint.h>

// ---------------------------------------------------------------------------
// Types for CDNA5 WMMA
// ---------------------------------------------------------------------------
typedef _Float16 half16 __attribute__((ext_vector_type(16)));
typedef _Float16 half8  __attribute__((ext_vector_type(8)));
typedef _Float16 half2v __attribute__((ext_vector_type(2)));
typedef float    float8 __attribute__((ext_vector_type(8)));

#define WMMA_F32_F16(A, B, C) \
    __builtin_amdgcn_wmma_f32_16x16x32_f16(false, (A), false, (B), (short)0, (C), false, false)

__device__ __forceinline__ half16 mk16(half8 lo, half8 hi) {
    return __builtin_shufflevector(lo, hi, 0,1,2,3,4,5,6,7,8,9,10,11,12,13,14,15);
}
__device__ __forceinline__ half16 load_frag(const _Float16* p) {
    // 16-bit A/B fragment: two contiguous half8 runs at +0 and +16 elements
    return mk16(*(const half8*)(p), *(const half8*)(p + 16));
}

// ---------------------------------------------------------------------------
// DPP16-based 16-lane reductions (stay within lane groups 0-15 / 16-31,
// matching the WMMA C/D fragment row groups). No LDS, full VALU rate.
// ---------------------------------------------------------------------------
template <int CTRL>
__device__ __forceinline__ float dpp_bcast(float x) {
    int t = __builtin_amdgcn_update_dpp(0, __builtin_bit_cast(int, x), CTRL, 0xF, 0xF, true);
    return __builtin_bit_cast(float, t);
}
__device__ __forceinline__ float row_max16(float x) {
    x = fmaxf(x, dpp_bcast<0xB1>(x));    // quad_perm [1,0,3,2]  (xor 1)
    x = fmaxf(x, dpp_bcast<0x4E>(x));    // quad_perm [2,3,0,1]  (xor 2)
    x = fmaxf(x, dpp_bcast<0x141>(x));   // ROW_HALF_MIRROR      (xor 7)
    x = fmaxf(x, dpp_bcast<0x140>(x));   // ROW_MIRROR           (xor 15)
    return x;
}
__device__ __forceinline__ float row_sum16(float x) {
    x += dpp_bcast<0xB1>(x);
    x += dpp_bcast<0x4E>(x);
    x += dpp_bcast<0x141>(x);
    x += dpp_bcast<0x140>(x);
    return x;
}

static constexpr int Bb = 4;
static constexpr int Tt = 2048;
static constexpr int Cc = 1024;
static constexpr int Hh = 16;
static constexpr int Dd = 64;
static constexpr float SCALE = 0.125f;   // 64^-0.5

// ---------------------------------------------------------------------------
// 1) LayerNorm (f32 in -> f16 out), one block (256 threads) per row of 1024
// ---------------------------------------------------------------------------
__global__ void ln_kernel(const float* __restrict__ x,
                          const float* __restrict__ gamma,
                          const float* __restrict__ beta,
                          _Float16* __restrict__ xn) {
    const int row = blockIdx.x;
    const int tid = threadIdx.x;
    const float4 v = ((const float4*)(x + (size_t)row * Cc))[tid];

    __shared__ float r1[256];
    __shared__ float r2[256];
    r1[tid] = v.x + v.y + v.z + v.w;
    r2[tid] = v.x * v.x + v.y * v.y + v.z * v.z + v.w * v.w;
    __syncthreads();
    for (int off = 128; off > 0; off >>= 1) {
        if (tid < off) { r1[tid] += r1[tid + off]; r2[tid] += r2[tid + off]; }
        __syncthreads();
    }
    const float mean = r1[0] * (1.0f / Cc);
    const float var  = r2[0] * (1.0f / Cc) - mean * mean;
    const float rs   = rsqrtf(var + 1e-5f);

    const float4 g = ((const float4*)gamma)[tid];
    const float4 b = ((const float4*)beta)[tid];
    _Float16* o = xn + (size_t)row * Cc + tid * 4;
    o[0] = (_Float16)((v.x - mean) * rs * g.x + b.x);
    o[1] = (_Float16)((v.y - mean) * rs * g.y + b.y);
    o[2] = (_Float16)((v.z - mean) * rs * g.z + b.z);
    o[3] = (_Float16)((v.w - mean) * rs * g.w + b.w);
}

// ---------------------------------------------------------------------------
// 2) f32 -> f16 conversion
// ---------------------------------------------------------------------------
__global__ void cvt_kernel(const float* __restrict__ s, _Float16* __restrict__ d, int n) {
    int i = blockIdx.x * blockDim.x + threadIdx.x;
    if (i < n) d[i] = (_Float16)s[i];
}

// ---------------------------------------------------------------------------
// 3) WMMA GEMM:  Out[m, o] = sum_c A[m,c] * W[o,c]   (A: [M,1024], W: [1024,1024])
//    Each wave owns a 32x64 output tile (2 M-tiles x 4 N-tiles, 8 accumulators).
//    Inline fragment loads: the scheduler clauses them and keeps several in
//    flight per WMMA without exceeding the VGPR budget (no spills).
//    MODE 0: f16 out at [B,h,T,d]   (Q, K)
//    MODE 1: f32 out row-major [M, 1024]  (final projection)
//    MODE 2: f16 out at [B,h,d,T]   (V, transposed for PV B-fragments)
// ---------------------------------------------------------------------------
template <int MODE>
__global__ void gemm16_kernel(const _Float16* __restrict__ A,
                              const _Float16* __restrict__ W,
                              _Float16* __restrict__ Oh,
                              float* __restrict__ Of) {
    const int lane = threadIdx.x & 31;
    const int ln16 = lane & 15;
    const int grp  = lane >> 4;
    const int kb   = grp * 8;

    const int wid = blockIdx.x * 8 + (threadIdx.x >> 5);
    const int m0  = (wid >> 4) * 32;   // 8192/32 = 256 M-tiles
    const int n0  = (wid & 15) * 64;   // 1024/64 = 16  N-super-tiles

    float8 acc[2][4] = {};

    const _Float16* arow0 = A + (size_t)(m0 + ln16) * Cc;
    const _Float16* arow1 = A + (size_t)(m0 + 16 + ln16) * Cc;

    for (int k = 0; k < Cc; k += 32) {
        const half16 a0 = load_frag(arow0 + k + kb);
        const half16 a1 = load_frag(arow1 + k + kb);
#pragma unroll
        for (int nt = 0; nt < 4; ++nt) {
            const _Float16* wrow = W + (size_t)(n0 + nt * 16 + ln16) * Cc;
            const half16 bfr = load_frag(wrow + k + kb);
            acc[0][nt] = WMMA_F32_F16(a0, bfr, acc[0][nt]);
            acc[1][nt] = WMMA_F32_F16(a1, bfr, acc[1][nt]);
        }
    }

#pragma unroll
    for (int mt = 0; mt < 2; ++mt) {
#pragma unroll
        for (int nt = 0; nt < 4; ++nt) {
#pragma unroll
            for (int r = 0; r < 8; ++r) {
                const int m = m0 + mt * 16 + r + 8 * grp;   // global row (b*T + t)
                const int o = n0 + nt * 16 + ln16;          // output channel
                const float val = acc[mt][nt][r];
                if (MODE == 1) {
                    Of[(size_t)m * Cc + o] = val;
                } else {
                    const int b  = m >> 11;       // T = 2048
                    const int t  = m & 2047;
                    const int hh = o >> 6;        // d = 64
                    const int dd = o & 63;
                    size_t idx;
                    if (MODE == 0) idx = (((size_t)(b * Hh + hh)) * Tt + t) * Dd + dd;
                    else           idx = (((size_t)(b * Hh + hh)) * Dd + dd) * Tt + t;
                    Oh[idx] = (_Float16)val;
                }
            }
        }
    }
}

// ---------------------------------------------------------------------------
// 4) Flash attention: one wave per (b, h, 16-query-row tile).
//    Q,K in [B,h,T,d] (f16), V in [B,h,d,T] (f16). Y out row-major [B*T, C] f16.
//
//    S sub-tiles are interleaved over keys: sub-tile st covers key
//    jb*32 + 2*ln16 + st, so each lane's two P values are adjacent in the
//    LDS transpose buffer and pack into a single b32 store.
// ---------------------------------------------------------------------------
__global__ void attn_kernel(const _Float16* __restrict__ Q,
                            const _Float16* __restrict__ K,
                            const _Float16* __restrict__ Vt,
                            _Float16* __restrict__ Y) {
    __shared__ __align__(16) _Float16 stage[4][16 * 32];

    const int lane = threadIdx.x & 31;
    const int ln16 = lane & 15;
    const int grp  = lane >> 4;
    const int kb   = grp * 8;
    const int wv   = threadIdx.x >> 5;

    const int wid   = blockIdx.x * 4 + wv;      // 8192 waves total
    const int qi    = wid & 127;                // 128 query tiles of 16
    const int bh    = wid >> 7;                 // b*16 + h
    const int b     = bh >> 4;
    const int hh    = bh & 15;
    const int qbase = qi * 16;

    const _Float16* Qh  = Q  + (size_t)bh * Tt * Dd;
    const _Float16* Kh  = K  + (size_t)bh * Tt * Dd;
    const _Float16* Vth = Vt + (size_t)bh * Dd * Tt;

    // Q A-fragments for the two K-dim (d) chunks of 32
    const _Float16* qrow = Qh + (size_t)(qbase + ln16) * Dd;
    const half16 qa0 = load_frag(qrow + 0  + kb);
    const half16 qa1 = load_frag(qrow + 32 + kb);

    float8 o_acc[4] = {};
    float  m_row[8], l_row[8];
#pragma unroll
    for (int r = 0; r < 8; ++r) { m_row[r] = -1e30f; l_row[r] = 0.0f; }

    for (int jb = 0; jb * 32 <= qbase + 15; ++jb) {
        // ---- S = Q K^T for 32 keys (two interleaved 16-col sub-tiles) ----
        float8 s0 = {}, s1 = {};
        {
            const _Float16* krow0 = Kh + (size_t)(jb * 32 + 2 * ln16 + 0) * Dd;
            const _Float16* krow1 = Kh + (size_t)(jb * 32 + 2 * ln16 + 1) * Dd;
            s0 = WMMA_F32_F16(qa0, load_frag(krow0 + 0  + kb), s0);
            s0 = WMMA_F32_F16(qa1, load_frag(krow0 + 32 + kb), s0);
            s1 = WMMA_F32_F16(qa0, load_frag(krow1 + 0  + kb), s1);
            s1 = WMMA_F32_F16(qa1, load_frag(krow1 + 32 + kb), s1);
        }

        // ---- scale + causal mask + online softmax (row stats via DPP16) ----
#pragma unroll
        for (int r = 0; r < 8; ++r) {
            const int row  = qbase + r + 8 * grp;
            const int col0 = jb * 32 + 2 * ln16;
            const int col1 = col0 + 1;
            float v0 = (col0 <= row) ? s0[r] * SCALE : -1e30f;
            float v1 = (col1 <= row) ? s1[r] * SCALE : -1e30f;

            const float mx   = row_max16(fmaxf(v0, v1));
            const float mnew = fmaxf(m_row[r], mx);

            const float p0 = __expf(v0 - mnew);
            const float p1 = __expf(v1 - mnew);
            const float ls = row_sum16(p0 + p1);

            const float alpha = __expf(m_row[r] - mnew);
            l_row[r] = l_row[r] * alpha + ls;
            m_row[r] = mnew;
            s0[r] = p0;
            s1[r] = p1;
            o_acc[0][r] *= alpha;
            o_acc[1][r] *= alpha;
            o_acc[2][r] *= alpha;
            o_acc[3][r] *= alpha;
        }

        // ---- transpose P (C/D layout -> A layout) through LDS ----
        _Float16* st = &stage[wv][0];
#pragma unroll
        for (int r = 0; r < 8; ++r) {
            const int row = r + 8 * grp;
            half2v pk;
            pk[0] = (_Float16)s0[r];
            pk[1] = (_Float16)s1[r];
            *(half2v*)(st + row * 32 + 2 * ln16) = pk;   // keys 2*ln16, 2*ln16+1
        }
        asm volatile("s_wait_dscnt 0" ::: "memory");
        const half16 pa = load_frag(st + ln16 * 32 + kb);

        // ---- O += P V  (V transposed: contiguous over key index) ----
#pragma unroll
        for (int nt = 0; nt < 4; ++nt) {
            const _Float16* vcol = Vth + (size_t)(nt * 16 + ln16) * Tt + jb * 32;
            o_acc[nt] = WMMA_F32_F16(pa, load_frag(vcol + kb), o_acc[nt]);
        }
    }

    // ---- normalize and store y row-major [B*T, C] ----
#pragma unroll
    for (int nt = 0; nt < 4; ++nt) {
#pragma unroll
        for (int r = 0; r < 8; ++r) {
            const int t = qbase + r + 8 * grp;
            const float yv = o_acc[nt][r] / l_row[r];
            const size_t idx = ((size_t)b * Tt + t) * Cc + hh * Dd + nt * 16 + ln16;
            Y[idx] = (_Float16)yv;
        }
    }
}

// ---------------------------------------------------------------------------
// Launch
// ---------------------------------------------------------------------------
extern "C" void kernel_launch(void* const* d_in, const int* in_sizes, int n_in,
                              void* d_out, int out_size, void* d_ws, size_t ws_size,
                              hipStream_t stream) {
    const float* x     = (const float*)d_in[0];
    const float* gamma = (const float*)d_in[1];
    const float* beta  = (const float*)d_in[2];
    const float* Wq    = (const float*)d_in[3];
    const float* Wk    = (const float*)d_in[4];
    const float* Wv    = (const float*)d_in[5];
    const float* Wo    = (const float*)d_in[6];
    float* out         = (float*)d_out;

    uint8_t* w = (uint8_t*)d_ws;
    const size_t MB = 1u << 20;
    _Float16* xn  = (_Float16*)(w + 0);         // 16 MB, reused as Y
    _Float16* Wqh = (_Float16*)(w + 16 * MB);   // 2 MB
    _Float16* Wkh = (_Float16*)(w + 18 * MB);
    _Float16* Wvh = (_Float16*)(w + 20 * MB);
    _Float16* Woh = (_Float16*)(w + 22 * MB);
    _Float16* Qb  = (_Float16*)(w + 24 * MB);   // 16 MB each
    _Float16* Kb  = (_Float16*)(w + 40 * MB);
    _Float16* Vt  = (_Float16*)(w + 56 * MB);   // end: 72 MB

    const int Mrows = Bb * Tt;                  // 8192
    const int wElems = Cc * Cc;                 // 1048576

    ln_kernel<<<Mrows, 256, 0, stream>>>(x, gamma, beta, xn);
    cvt_kernel<<<wElems / 256, 256, 0, stream>>>(Wq, Wqh, wElems);
    cvt_kernel<<<wElems / 256, 256, 0, stream>>>(Wk, Wkh, wElems);
    cvt_kernel<<<wElems / 256, 256, 0, stream>>>(Wv, Wvh, wElems);
    cvt_kernel<<<wElems / 256, 256, 0, stream>>>(Wo, Woh, wElems);

    const int gemmBlocks = (Mrows / 32) * 16 / 8;   // 512 blocks of 256 (8 waves)
    gemm16_kernel<0><<<gemmBlocks, 256, 0, stream>>>(xn, Wqh, Qb, nullptr);
    gemm16_kernel<0><<<gemmBlocks, 256, 0, stream>>>(xn, Wkh, Kb, nullptr);
    gemm16_kernel<2><<<gemmBlocks, 256, 0, stream>>>(xn, Wvh, Vt, nullptr);

    attn_kernel<<<(Bb * Hh * (Tt / 16)) / 4, 128, 0, stream>>>(Qb, Kb, Vt, xn /* Y */);

    gemm16_kernel<1><<<gemmBlocks, 256, 0, stream>>>(xn /* Y */, Woh, nullptr, out);
}
